// Model_28681791602767
// MI455X (gfx1250) — matrix operand order repaired and checked by reading, not compiled, not run on hardware
//
#include <hip/hip_runtime.h>

// CDNA5 / gfx1250, wave32 only.
typedef __attribute__((ext_vector_type(16))) __bf16 v16bf;
typedef __attribute__((ext_vector_type(8)))  __bf16 v8bf;
typedef __attribute__((ext_vector_type(8)))  float  v8f;
typedef __attribute__((ext_vector_type(4)))  float  v4f;

#define B_   2
#define H_   16
#define N_   4096
#define D_   128
#define E_   128
#define BLK  64
#define NBLK 64
#define KP   136   // bf16 elements per kvT row (128 + 8 pad: 16B-aligned rows, spreads banks)

__global__ __launch_bounds__(256) void linattn_nondiag(
    const float* __restrict__ q,     // [B,H,N,D]
    const float* __restrict__ outp,  // [B,H,N,E]
    const float* __restrict__ s,     // [H]
    const float* __restrict__ kv,    // [B,H,NBLK,D,E]
    float* __restrict__ out)         // [B,H,N,E]
{
    extern __shared__ char smem_raw[];
    __bf16* kvh = (__bf16*)smem_raw;        // kv^T hi plane: [E][KP]
    __bf16* kvl = kvh + E_ * KP;            // kv^T lo plane: [E][KP]

    const int flat = blockIdx.x;            // 0 .. B*H*NBLK-1
    const int nb   = flat & (NBLK - 1);
    const int bh   = flat >> 6;             // b*H + h
    const int h    = bh & (H_ - 1);

    const float  sh     = s[h];
    const size_t qbase  = ((size_t)bh * N_ + (size_t)nb * BLK) * D_;
    const size_t kvbase = ((size_t)bh * NBLK + nb) * (size_t)(D_ * E_);
    const size_t obase  = ((size_t)bh * N_ + (size_t)nb * BLK) * E_;

    const int tid = threadIdx.x;

    // ---- Stage kv^T into LDS as bf16 hi/lo (split for compensated precision) ----
    {
        const v4f* kv4 = (const v4f*)(kv + kvbase);   // 4096 float4s
        #pragma unroll
        for (int i = 0; i < 16; ++i) {
            int f4 = i * 256 + tid;
            int d  = f4 >> 5;            // row (D index)
            int e0 = (f4 & 31) * 4;      // col (E index)
            v4f v  = kv4[f4];
            #pragma unroll
            for (int j = 0; j < 4; ++j) {
                float  x  = v[j];
                __bf16 hi = (__bf16)x;
                float  rr = x - (float)hi;
                kvh[(e0 + j) * KP + d] = hi;
                kvl[(e0 + j) * KP + d] = (__bf16)rr;
            }
        }
    }
    __syncthreads();

    // ---- Per-wave tile assignment: wave w -> m-tile (w&3), n-group (w>>2) ----
    const int lane    = tid & 31;
    const int w       = tid >> 5;
    const int tm      = w & 3;           // 16-row m-tile within the 64-row block
    const int ng      = w >> 2;          // 0 or 1 -> 64-col group
    const int colbase = ng * 64;

    // A-matrix 16x32 bf16 layout: lanes 0-15 hold rows 0-15 with K = {0..7,16..23},
    // lanes 16-31 hold rows 0-15 with K = {8..15,24..31}.
    const int   rloc  = tm * 16 + (lane & 15);        // q row within block (decay index)
    const float decay = __expf(-sh * (float)rloc);    // fold decay into q rows
    const float* qrow = q + qbase + (size_t)rloc * D_;
    const int kb  = (lane >> 4) * 8;                  // A sub-K offset
    const int kb2 = (lane >> 4) * 8;                  // B sub-K offset
    const int bcol = lane & 15;                       // B column (N) within tile

    v8f acc[4] = {};

    #pragma unroll
    for (int kc = 0; kc < 4; ++kc) {                  // K chunks of 32
        const int k0 = kc * 32 + kb;
        v4f a0 = *(const v4f*)(qrow + k0);
        v4f a1 = *(const v4f*)(qrow + k0 + 4);
        v4f a2 = *(const v4f*)(qrow + k0 + 16);
        v4f a3 = *(const v4f*)(qrow + k0 + 20);

        v16bf ah, al;
        #pragma unroll
        for (int j = 0; j < 4; ++j) {
            float x, fh;
            x = a0[j] * decay; ah[j]      = (__bf16)x; fh = (float)ah[j];      al[j]      = (__bf16)(x - fh);
            x = a1[j] * decay; ah[4 + j]  = (__bf16)x; fh = (float)ah[4 + j];  al[4 + j]  = (__bf16)(x - fh);
            x = a2[j] * decay; ah[8 + j]  = (__bf16)x; fh = (float)ah[8 + j];  al[8 + j]  = (__bf16)(x - fh);
            x = a3[j] * decay; ah[12 + j] = (__bf16)x; fh = (float)ah[12 + j]; al[12 + j] = (__bf16)(x - fh);
        }

        #pragma unroll
        for (int nt = 0; nt < 4; ++nt) {
            const int ecol = colbase + nt * 16 + bcol;          // kv^T row = E index
            const __bf16* bph = kvh + (size_t)ecol * KP + kc * 32 + kb2;
            const __bf16* bpl = kvl + (size_t)ecol * KP + kc * 32 + kb2;
            // 16B-aligned halves -> two ds_load_b128 each
            v8bf bh0 = *(const v8bf*)(bph);
            v8bf bh1 = *(const v8bf*)(bph + 16);
            v8bf bl0 = *(const v8bf*)(bpl);
            v8bf bl1 = *(const v8bf*)(bpl + 16);
            v16bf bhv = __builtin_shufflevector(bh0, bh1, 0,1,2,3,4,5,6,7,8,9,10,11,12,13,14,15);
            v16bf blv = __builtin_shufflevector(bl0, bl1, 0,1,2,3,4,5,6,7,8,9,10,11,12,13,14,15);

            // q*kv = qh*kh + ql*kh + qh*kl  (ql*kl ~ 2^-16, dropped)
            acc[nt] = __builtin_amdgcn_wmma_f32_16x16x32_bf16(
                false, ah, false, bhv, (short)0, acc[nt], false, false);
            acc[nt] = __builtin_amdgcn_wmma_f32_16x16x32_bf16(
                false, al, false, bhv, (short)0, acc[nt], false, false);
            acc[nt] = __builtin_amdgcn_wmma_f32_16x16x32_bf16(
                false, ah, false, blv, (short)0, acc[nt], false, false);
        }
    }

    // ---- Epilogue: out = output + acc. C/D layout: VGPR j -> row j (lanes 0-15) / j+8 ----
    const float* op = outp + obase;
    float*       od = out + obase;
    const int rb = tm * 16 + ((lane >> 4) << 3);
    #pragma unroll
    for (int nt = 0; nt < 4; ++nt) {
        const int colg = colbase + nt * 16 + (lane & 15);
        #pragma unroll
        for (int j = 0; j < 8; ++j) {
            const size_t idx = (size_t)(rb + j) * E_ + colg;
            od[idx] = op[idx] + acc[nt][j];
        }
    }
}

extern "C" void kernel_launch(void* const* d_in, const int* in_sizes, int n_in,
                              void* d_out, int out_size, void* d_ws, size_t ws_size,
                              hipStream_t stream) {
    const float* q    = (const float*)d_in[0];   // [B,H,N,D] f32
    const float* outp = (const float*)d_in[1];   // [B,H,N,E] f32
    const float* s    = (const float*)d_in[2];   // [H] f32
    const float* kv   = (const float*)d_in[3];   // [B,H,NBLK,D,E] f32
    float*       out  = (float*)d_out;

    const size_t smem = (size_t)2 * E_ * KP * sizeof(__bf16);  // hi+lo kv^T planes (~68 KB of 320 KB LDS)
    const dim3 grid(B_ * H_ * NBLK);   // 2048 workgroups
    const dim3 block(256);             // 8 wave32s
    linattn_nondiag<<<grid, block, smem, stream>>>(q, outp, s, kv, out);
}
